// MultiHeadAttention_42468636623502
// MI455X (gfx1250) — matrix operand order repaired and checked by reading, not compiled, not run on hardware
//
#include <hip/hip_runtime.h>
#include <hip/hip_bf16.h>

typedef __attribute__((ext_vector_type(16))) __bf16 v16bf;
typedef __attribute__((ext_vector_type(8)))  __bf16 v8bf;
typedef __attribute__((ext_vector_type(4)))  __bf16 v4bf;
typedef __attribute__((ext_vector_type(8)))  float  v8f;

#define D_MODEL 1024
#define NHEAD 16
#define DK 64
#define BATCH 4
#define SEQ 2048

union V16U { v16bf v; v8bf h[2]; };

// A-fragment (16x32, 16-bit): per lane, elements 0..15 = k[16*half .. 16*half+15]
// (contiguous, matching B fragments) -> two 16B loads.
__device__ __forceinline__ v16bf load_a_frag(const __bf16* rowbase, int half) {
    V16U u;
    u.h[0] = *reinterpret_cast<const v8bf*>(rowbase + 16 * half);
    u.h[1] = *reinterpret_cast<const v8bf*>(rowbase + 16 * half + 8);
    return u.v;
}

// Async DMA: global -> LDS, 16 bytes, tracked by ASYNCcnt (CDNA5 §10 / §15.18).
__device__ __forceinline__ void async_copy_b128(const __bf16* lds_dst, const __bf16* g_src) {
    unsigned lds_off = (unsigned)(unsigned long long)lds_dst;   // flat LDS: addr[31:0]
    unsigned long long ga = (unsigned long long)g_src;
    asm volatile("global_load_async_to_lds_b128 %0, %1, off"
                 :: "v"(lds_off), "v"(ga) : "memory");
}
__device__ __forceinline__ void wait_async_le(int n) {
    if (n == 0)      asm volatile("s_wait_asynccnt 0x0" ::: "memory");
    else if (n == 1) asm volatile("s_wait_asynccnt 0x1" ::: "memory");
    else             asm volatile("s_wait_asynccnt 0x2" ::: "memory");
}

// reductions across the 16 lanes of a wave-half (rows of a C fragment)
__device__ __forceinline__ float half16_max(float v) {
#pragma unroll
    for (int off = 1; off < 16; off <<= 1)
        v = fmaxf(v, __shfl_xor(v, off, 32));
    return v;
}
__device__ __forceinline__ float half16_sum(float v) {
#pragma unroll
    for (int off = 1; off < 16; off <<= 1)
        v += __shfl_xor(v, off, 32);
    return v;
}

// ---------------------------------------------------------------------------
// Weight prep: straight f32 -> bf16 convert (torch W[n][k] layout is already
// the n-major/k-contiguous layout the B fragments want).
// ---------------------------------------------------------------------------
__global__ __launch_bounds__(256) void convert_to_bf16(
        const float* __restrict__ src, __bf16* __restrict__ dst) {
    int i = (blockIdx.x * 256 + threadIdx.x) * 4;
    float4 f = *reinterpret_cast<const float4*>(src + i);
    v4bf o;
    o[0] = (__bf16)f.x; o[1] = (__bf16)f.y; o[2] = (__bf16)f.z; o[3] = (__bf16)f.w;
    *reinterpret_cast<v4bf*>(dst + i) = o;
}

// ---------------------------------------------------------------------------
// C[M,1024] = A[M,1024](f32) @ W^T + bias, W given bf16 in [n][k] layout.
// Block tile 128(M) x 64(N), 8 waves, k-step 32. B tile double-buffered via
// async-to-LDS DMA; A tile staged synchronously (needs f32->bf16 convert).
// mode 0: f32 row-major | mode 1: bf16 [b][h][s][64] | mode 2: bf16 [b][h][64][s]
// ---------------------------------------------------------------------------
__global__ __launch_bounds__(256) void gemm_xwt(
        const float* __restrict__ A, const __bf16* __restrict__ Wb,
        const float* __restrict__ bias, void* __restrict__ out, int mode) {
    __shared__ __bf16 As[128 * 32];       // m-major, k contiguous
    __shared__ __bf16 Bs[2][64 * 32];     // n-major, k contiguous (double buffer)
    const int tid  = threadIdx.x;
    const int wave = tid >> 5, lane = tid & 31;
    const int half = lane >> 4, ln = lane & 15;
    const int m0 = blockIdx.x * 128;
    const int n0 = blockIdx.y * 64;

    const int brow = tid >> 2, bcol = (tid & 3) * 8;
    const __bf16* bsrc = Wb + (size_t)(n0 + brow) * 1024 + bcol;
    async_copy_b128(&Bs[0][brow * 32 + bcol], bsrc);          // prologue: B tile 0

    v8f acc[4] = {};
    for (int kk = 0; kk < 1024; kk += 32) {
        const int cur = (kk >> 5) & 1;
        {   // stage A tile 128x32 (convert f32 -> bf16), vectorized both sides
            int row = tid >> 1;
            int col = (tid & 1) * 16;
            const float* ap = A + (size_t)(m0 + row) * 1024 + kk + col;
            __bf16* dp = &As[row * 32 + col];
#pragma unroll
            for (int v = 0; v < 4; ++v) {
                float4 f = reinterpret_cast<const float4*>(ap)[v];
                v4bf o;
                o[0] = (__bf16)f.x; o[1] = (__bf16)f.y;
                o[2] = (__bf16)f.z; o[3] = (__bf16)f.w;
                *reinterpret_cast<v4bf*>(dp + 4 * v) = o;
            }
            if (kk + 32 < 1024) __builtin_prefetch(ap + 32, 0, 3);
        }
        if (kk + 32 < 1024) {             // issue next B tile into other buffer
            async_copy_b128(&Bs[cur ^ 1][brow * 32 + bcol], bsrc + kk + 32);
            wait_async_le(1);             // current B tile resident, next in flight
        } else {
            wait_async_le(0);
        }
        __syncthreads();

        v16bf af = load_a_frag(&As[(wave * 16 + ln) * 32], half);
#pragma unroll
        for (int j = 0; j < 4; ++j) {
            const v16bf bfrag = *reinterpret_cast<const v16bf*>(
                &Bs[cur][(j * 16 + ln) * 32 + 16 * half]);
            acc[j] = __builtin_amdgcn_wmma_f32_16x16x32_bf16(
                false, af, false, bfrag, (short)0, acc[j], false, false);
        }
        __syncthreads();
    }
    // epilogue: bias add + store
#pragma unroll
    for (int j = 0; j < 4; ++j) {
        int ng = n0 + j * 16 + ln;
        float bv = bias[ng];
#pragma unroll
        for (int r = 0; r < 8; ++r) {
            int mg = m0 + wave * 16 + r + 8 * half;
            float val = acc[j][r] + bv;
            if (mode == 0) {
                ((float*)out)[(size_t)mg * 1024 + ng] = val;
            } else {
                int b = mg >> 11, s = mg & 2047;
                int h = ng >> 6,  d = ng & 63;
                if (mode == 1)
                    ((__bf16*)out)[(((size_t)b * NHEAD + h) * SEQ + s) * DK + d] = (__bf16)val;
                else
                    ((__bf16*)out)[(((size_t)b * NHEAD + h) * DK + d) * SEQ + s] = (__bf16)val;
            }
        }
    }
}

// ---------------------------------------------------------------------------
// Flash attention, causal. Grid: (SEQ/128, BATCH*NHEAD), 256 threads.
// q/k in [b][h][s][64] bf16; v in [b][h][64][s] bf16 (pre-transposed).
// K/V tiles double-buffered via async-to-LDS DMA (2 ops per tile in flight).
// ---------------------------------------------------------------------------
__global__ __launch_bounds__(256) void flash_attn(
        const __bf16* __restrict__ qh, const __bf16* __restrict__ kh,
        const __bf16* __restrict__ vt, float* __restrict__ outc) {
    __shared__ __bf16 Kt[2][32 * 64];     // key-major, d contiguous (for Q.K^T)
    __shared__ __bf16 Vs[2][64 * 32];     // d-major, key contiguous (for P.V)
    __shared__ __bf16 Pt[8 * 16 * 32];    // per-wave P relayout scratch
    const int tid  = threadIdx.x;
    const int wave = tid >> 5, lane = tid & 31;
    const int half = lane >> 4, ln = lane & 15;
    const int bh = blockIdx.y;
    const int b = bh >> 4, h = bh & 15;
    const int q_blk = blockIdx.x * 128;
    const int q0 = q_blk + wave * 16;

    const __bf16* Qb = qh + (size_t)bh * SEQ * DK;
    const __bf16* Kb = kh + (size_t)bh * SEQ * DK;
    const __bf16* Vb = vt + (size_t)bh * DK * SEQ;

    // Q fragments for this wave's 16 rows (d = 0..31, 32..63)
    v16bf qa[2];
#pragma unroll
    for (int f = 0; f < 2; ++f)
        qa[f] = load_a_frag(Qb + (size_t)(q0 + ln) * DK + 32 * f, half);

    v8f o[4] = {};
    float mrow[8], lrow[8];
#pragma unroll
    for (int r = 0; r < 8; ++r) { mrow[r] = -1e30f; lrow[r] = 0.0f; }

    const int krow = tid >> 3, kcol = (tid & 7) * 8;   // K: 32x64 copy slice
    const int vrow = tid >> 2, vcol = (tid & 3) * 8;   // V: 64x32 copy slice
    const __bf16* ksrc = Kb + (size_t)krow * DK + kcol;
    const __bf16* vsrc = Vb + (size_t)vrow * SEQ + vcol;

    __bf16* Pw = &Pt[wave * 512];
    const int t_cnt = (q_blk + 128) / 32;              // causal bound for block

    // prologue: tile 0 into buffer 0 (two async b128 per thread)
    async_copy_b128(&Kt[0][krow * 64 + kcol], ksrc);
    async_copy_b128(&Vs[0][vrow * 32 + vcol], vsrc);

    for (int t = 0; t < t_cnt; ++t) {
        const int k0 = t * 32;
        const int cur = t & 1;
        if (t + 1 < t_cnt) {              // issue next K/V tiles into other buffer
            async_copy_b128(&Kt[cur ^ 1][krow * 64 + kcol], ksrc + (size_t)(k0 + 32) * DK);
            async_copy_b128(&Vs[cur ^ 1][vrow * 32 + vcol], vsrc + (k0 + 32));
            wait_async_le(2);             // tile t resident; tile t+1 (2 ops) in flight
        } else {
            wait_async_le(0);
        }
        __syncthreads();

        if (k0 <= q0 + 15) {              // wave-uniform: EXEC stays full
            // scores S = Q Ktile^T  (16 x 32)
            v8f sc[2];
#pragma unroll
            for (int nt = 0; nt < 2; ++nt) {
                v8f c = {};
#pragma unroll
                for (int f = 0; f < 2; ++f) {
                    const v16bf kbf = *reinterpret_cast<const v16bf*>(
                        &Kt[cur][(nt * 16 + ln) * 64 + 32 * f + 16 * half]);
                    c = __builtin_amdgcn_wmma_f32_16x16x32_bf16(
                        false, qa[f], false, kbf, (short)0, c, false, false);
                }
                sc[nt] = c;
            }
            // online softmax update
#pragma unroll
            for (int r = 0; r < 8; ++r) {
                int qr = q0 + r + 8 * half;
                float v0 = sc[0][r] * 0.125f;     // 1/sqrt(64)
                float v1 = sc[1][r] * 0.125f;
                if (k0 + ln      > qr) v0 = -1e9f;
                if (k0 + 16 + ln > qr) v1 = -1e9f;
                float mn = fmaxf(mrow[r], half16_max(fmaxf(v0, v1)));
                float alpha = __expf(mrow[r] - mn);
                float p0 = __expf(v0 - mn);
                float p1 = __expf(v1 - mn);
                lrow[r] = lrow[r] * alpha + half16_sum(p0 + p1);
                mrow[r] = mn;
                Pw[(r + 8 * half) * 32 + ln]      = (__bf16)p0;
                Pw[(r + 8 * half) * 32 + 16 + ln] = (__bf16)p1;
#pragma unroll
                for (int j = 0; j < 4; ++j) o[j][r] *= alpha;
            }
            asm volatile("s_wait_dscnt 0" ::: "memory");
            // O += P @ Vtile
            v16bf pa = load_a_frag(&Pw[ln * 32], half);
#pragma unroll
            for (int j = 0; j < 4; ++j) {
                const v16bf vbf = *reinterpret_cast<const v16bf*>(
                    &Vs[cur][(j * 16 + ln) * 32 + 16 * half]);
                o[j] = __builtin_amdgcn_wmma_f32_16x16x32_bf16(
                    false, pa, false, vbf, (short)0, o[j], false, false);
            }
        }
        __syncthreads();
    }
    // normalize + write concat layout [b][s][h*64+d], f32
#pragma unroll
    for (int j = 0; j < 4; ++j)
#pragma unroll
        for (int r = 0; r < 8; ++r) {
            int qg = q0 + r + 8 * half;
            float val = o[j][r] / lrow[r];
            outc[(size_t)(b * SEQ + qg) * D_MODEL + h * DK + j * 16 + ln] = val;
        }
}

// ---------------------------------------------------------------------------
extern "C" void kernel_launch(void* const* d_in, const int* in_sizes, int n_in,
                              void* d_out, int out_size, void* d_ws, size_t ws_size,
                              hipStream_t stream) {
    const float* Q    = (const float*)d_in[0];
    const float* K    = (const float*)d_in[1];
    const float* V    = (const float*)d_in[2];
    // d_in[3] = mask (causal; recomputed analytically)
    const float* wq_w = (const float*)d_in[4];
    const float* wq_b = (const float*)d_in[5];
    const float* wk_w = (const float*)d_in[6];
    const float* wk_b = (const float*)d_in[7];
    const float* wv_w = (const float*)d_in[8];
    const float* wv_b = (const float*)d_in[9];
    const float* wo_w = (const float*)d_in[10];
    const float* wo_b = (const float*)d_in[11];

    char* ws = (char*)d_ws;
    const size_t WBYTES = (size_t)D_MODEL * D_MODEL * sizeof(__bf16);   // 2 MB
    __bf16* wqb = (__bf16*)(ws + 0 * WBYTES);
    __bf16* wkb = (__bf16*)(ws + 1 * WBYTES);
    __bf16* wvb = (__bf16*)(ws + 2 * WBYTES);
    __bf16* wob = (__bf16*)(ws + 3 * WBYTES);
    size_t off = 4 * WBYTES;
    const size_t HBYTES = (size_t)BATCH * SEQ * D_MODEL * sizeof(__bf16); // 16 MB
    __bf16* qh = (__bf16*)(ws + off); off += HBYTES;
    __bf16* kh = (__bf16*)(ws + off); off += HBYTES;
    __bf16* vt = (__bf16*)(ws + off); off += HBYTES;
    float* attnc = (float*)(ws + off);                                   // 32 MB

    dim3 tb(256);
    int cgrid = (D_MODEL * D_MODEL) / (256 * 4);
    convert_to_bf16<<<cgrid, tb, 0, stream>>>(wq_w, wqb);
    convert_to_bf16<<<cgrid, tb, 0, stream>>>(wk_w, wkb);
    convert_to_bf16<<<cgrid, tb, 0, stream>>>(wv_w, wvb);
    convert_to_bf16<<<cgrid, tb, 0, stream>>>(wo_w, wob);

    dim3 gg((BATCH * SEQ) / 128, D_MODEL / 64);
    gemm_xwt<<<gg, tb, 0, stream>>>(Q, wqb, wq_b, qh, 1);
    gemm_xwt<<<gg, tb, 0, stream>>>(K, wkb, wk_b, kh, 1);
    gemm_xwt<<<gg, tb, 0, stream>>>(V, wvb, wv_b, vt, 2);

    flash_attn<<<dim3(SEQ / 128, BATCH * NHEAD), tb, 0, stream>>>(qh, kh, vt, attnc);

    gemm_xwt<<<gg, tb, 0, stream>>>(attnc, wob, wo_b, (float*)d_out, 0);
}
